// GCN4_19808389169217
// MI455X (gfx1250) — compile-verified, matrix operand order checked by reading, not run on hardware
//
#include <hip/hip_runtime.h>
#include <hip/hip_bf16.h>

typedef __attribute__((ext_vector_type(16))) __bf16 v16bf;
typedef __attribute__((ext_vector_type(8)))  float  v8f;

#define WAVES_PER_BLOCK 4

// ---------------------------------------------------------------------------
// Pack W[K x 16] (f32 row-major) into per-(kstep,lane) WMMA B fragments:
// Bp[(kk*32+lane)*16 + s] = bf16(W[k*16 + col]) with the ISA 16-bit B layout
//   lane<16 : col N=lane,   slots 0..7 -> K=kbase+0..7,  8..15 -> kbase+16..23
//   lane>=16: col N=lane-16,slots 0..7 -> K=kbase+8..15, 8..15 -> kbase+24..31
// ---------------------------------------------------------------------------
template <int K_TOTAL>
__global__ void k_pack_b(const float* __restrict__ W, __bf16* __restrict__ Bp)
{
    constexpr int KSTEPS = (K_TOTAL + 31) / 32;
    int t = threadIdx.x;
    if (t >= 32 * KSTEPS) return;
    int lane = t & 31, kk = t >> 5;
    int half = lane >> 4, l16 = lane & 15;
    int kbase = kk * 32;
#pragma unroll
    for (int s = 0; s < 16; ++s) {
        int k = kbase + ((s < 8) ? (8 * half + s) : (16 + 8 * half + (s - 8)));
        float v = (k < K_TOTAL) ? W[k * 16 + l16] : 0.0f;
        Bp[(size_t)t * 16 + s] = (__bf16)v;
    }
}

// ---------------------------------------------------------------------------
// Tall-skinny GEMM  Y[N x 16] = X[N x K] * W[K x 16]  via v_wmma_f32_16x16x32_bf16
// One wave per 16-row tile. A loaded as 4x float4 per k-step (contiguous 8-float
// runs per lane half), converted via paired bf16 cvt; B fragment = one 32B
// coalesced load from the pre-packed buffer.
// ---------------------------------------------------------------------------
template <int K_TOTAL>
__global__ __launch_bounds__(WAVES_PER_BLOCK * 32)
void gcn_gemm_n16_wmma(const float* __restrict__ X,
                       const __bf16* __restrict__ Bpacked,
                       float* __restrict__ Y,
                       int nRows, int nTiles)
{
    const int lane = threadIdx.x & 31;
    const int wave = threadIdx.x >> 5;
    const int tile = blockIdx.x * WAVES_PER_BLOCK + wave;
    if (tile >= nTiles) return;                 // wave-uniform

    const int half = lane >> 4;                 // 0 or 1
    const int l16  = lane & 15;
    int row = tile * 16 + l16;
    if (row >= nRows) row = nRows - 1;          // clamp: no EXEC divergence
    const float4* __restrict__ x4 = (const float4*)(X + (size_t)row * K_TOTAL);
    const v16bf*  __restrict__ bp = (const v16bf*)Bpacked;

    constexpr int KSTEPS = (K_TOTAL + 31) / 32;
    v8f c = {};
#pragma unroll
    for (int kk = 0; kk < KSTEPS; ++kk) {
        const int kbase = kk * 32;
        v16bf a;
        {   // slots 0..7 : k = kbase + 8*half + {0..7}  (always < K_TOTAL)
            const int i0 = (kbase >> 2) + 2 * half;
            float4 a0 = x4[i0];
            float4 a1 = x4[i0 + 1];
            a[0] = (__bf16)a0.x; a[1] = (__bf16)a0.y;
            a[2] = (__bf16)a0.z; a[3] = (__bf16)a0.w;
            a[4] = (__bf16)a1.x; a[5] = (__bf16)a1.y;
            a[6] = (__bf16)a1.z; a[7] = (__bf16)a1.w;
        }
        if (kbase + 32 <= K_TOTAL) {            // compile-time after unroll
            // slots 8..15 : k = kbase + 16 + 8*half + {0..7}
            const int i2 = ((kbase + 16) >> 2) + 2 * half;
            float4 a2 = x4[i2];
            float4 a3 = x4[i2 + 1];
            a[8]  = (__bf16)a2.x; a[9]  = (__bf16)a2.y;
            a[10] = (__bf16)a2.z; a[11] = (__bf16)a2.w;
            a[12] = (__bf16)a3.x; a[13] = (__bf16)a3.y;
            a[14] = (__bf16)a3.z; a[15] = (__bf16)a3.w;
        } else {
#pragma unroll
            for (int s = 8; s < 16; ++s) a[s] = (__bf16)0.0f;
        }
        v16bf b = bp[kk * 32 + lane];           // 32B coalesced fragment
        c = __builtin_amdgcn_wmma_f32_16x16x32_bf16(
                /*neg_a=*/false, a, /*neg_b=*/false, b,
                /*c_mod=*/(short)0, c, /*reuse_a=*/false, /*reuse_b=*/false);
    }

    // D: VGPR j -> row tile*16 + j + half*8, col = l16 (coalesced 64B rows)
    if (tile * 16 + 16 <= nRows) {              // wave-uniform fast path
#pragma unroll
        for (int j = 0; j < 8; ++j) {
            const int m = tile * 16 + j + half * 8;
            Y[(size_t)m * 16 + l16] = c[j];
        }
    } else {
#pragma unroll
        for (int j = 0; j < 8; ++j) {
            const int m = tile * 16 + j + half * 8;
            if (m < nRows) Y[(size_t)m * 16 + l16] = c[j];
        }
    }
}

// ---------------------------------------------------------------------------
// Degree / normalization
// ---------------------------------------------------------------------------
__global__ void k_init_deg(float* deg, int n)
{
    int i = blockIdx.x * blockDim.x + threadIdx.x;
    if (i < n) deg[i] = 1.0f;                   // self-loop
}

__global__ void k_deg_edges(const int* __restrict__ dst, float* deg, int nE)
{
    int e = blockIdx.x * blockDim.x + threadIdx.x;
    if (e < nE) atomicAdd(&deg[dst[e]], 1.0f);
}

__global__ void k_rsqrt(float* deg, int n)      // in place: deg -> dinv
{
    int i = blockIdx.x * blockDim.x + threadIdx.x;
    if (i < n) deg[i] = rsqrtf(deg[i]);
}

__global__ void k_zero(float* p, size_t n)
{
    size_t i = (size_t)blockIdx.x * blockDim.x + threadIdx.x;
    if (i < n) p[i] = 0.0f;
}

// ---------------------------------------------------------------------------
// Edge aggregation: agg[dst] += xw[src] * dinv[src]*dinv[dst]
// 4 threads per edge, each handling one float4 of the 16-float row.
// ---------------------------------------------------------------------------
__global__ void k_edge_agg(const float4* __restrict__ xw4,
                           const float* __restrict__ dinv,
                           const int* __restrict__ src,
                           const int* __restrict__ dst,
                           float* agg, int nE)
{
    int t = blockIdx.x * blockDim.x + threadIdx.x;
    int e = t >> 2, q = t & 3;
    if (e >= nE) return;
    int s = src[e], d = dst[e];
    float w = dinv[s] * dinv[d];
    float4 v = xw4[(size_t)s * 4 + q];
    float* o = agg + (size_t)d * 16 + q * 4;
    atomicAdd(o + 0, v.x * w);
    atomicAdd(o + 1, v.y * w);
    atomicAdd(o + 2, v.z * w);
    atomicAdd(o + 3, v.w * w);
}

// x = relu(agg + xw*dinv^2 + b), written in place into agg
__global__ void k_finalize(float* agg, const float* __restrict__ xw,
                           const float* __restrict__ dinv,
                           const float* __restrict__ bias, size_t n)
{
    size_t i = (size_t)blockIdx.x * blockDim.x + threadIdx.x;
    if (i >= n) return;
    int node = (int)(i >> 4), h = (int)(i & 15);
    float di = dinv[node];
    float v = agg[i] + xw[i] * di * di + bias[h];
    agg[i] = v > 0.0f ? v : 0.0f;
}

// ---------------------------------------------------------------------------
// Global mean pool (atomics into Gx16 sums + G counts)
// ---------------------------------------------------------------------------
__global__ void k_pool(const float* __restrict__ x,
                       const int* __restrict__ batch,
                       float* sums, float* cnts, size_t n)
{
    size_t i = (size_t)blockIdx.x * blockDim.x + threadIdx.x;
    if (i >= n) return;
    int node = (int)(i >> 4), h = (int)(i & 15);
    int g = batch[node];
    atomicAdd(&sums[g * 16 + h], x[i]);
    if (h == 0) atomicAdd(&cnts[g], 1.0f);
}

// head: pooled = sums/max(cnt,1); h = relu(pooled@Wf+bf); out = h@Wo+bo
__global__ void k_head(const float* __restrict__ sums,
                       const float* __restrict__ cnts,
                       const float* __restrict__ Wf, const float* __restrict__ bf_,
                       const float* __restrict__ Wo, const float* __restrict__ bo,
                       float* out, int nG)
{
    int g = blockIdx.x * blockDim.x + threadIdx.x;
    if (g >= nG) return;
    float cnt = fmaxf(cnts[g], 1.0f);
    float p[16];
#pragma unroll
    for (int h = 0; h < 16; ++h) p[h] = sums[g * 16 + h] / cnt;
    float hv[16];
#pragma unroll
    for (int j = 0; j < 16; ++j) {
        float s = bf_[j];
#pragma unroll
        for (int k = 0; k < 16; ++k) s += p[k] * Wf[k * 16 + j];
        hv[j] = fmaxf(s, 0.0f);
    }
#pragma unroll
    for (int o = 0; o < 2; ++o) {
        float s = bo[o];
#pragma unroll
        for (int k = 0; k < 16; ++k) s += hv[k] * Wo[k * 2 + o];
        out[g * 2 + o] = s;
    }
}

// ---------------------------------------------------------------------------
extern "C" void kernel_launch(void* const* d_in, const int* in_sizes, int n_in,
                              void* d_out, int out_size, void* d_ws, size_t ws_size,
                              hipStream_t stream)
{
    const float* features = (const float*)d_in[0];
    const float* W1 = (const float*)d_in[1];
    const float* b1 = (const float*)d_in[2];
    const float* W2 = (const float*)d_in[3];
    const float* b2 = (const float*)d_in[4];
    const float* Wf = (const float*)d_in[5];
    const float* bf_ = (const float*)d_in[6];
    const float* Wo = (const float*)d_in[7];
    const float* bo = (const float*)d_in[8];
    const int* src   = (const int*)d_in[9];
    const int* dst   = (const int*)d_in[10];
    const int* batch = (const int*)d_in[11];

    const int N = in_sizes[11];          // 100000 nodes
    const int E = in_sizes[9];           // 1600000 edges
    const int G = out_size / 2;          // 256 graphs
    const size_t NF = (size_t)N * 16;

    // workspace layout (floats), aligned regions
    float* base = (float*)d_ws;
    size_t npad = ((size_t)N + 255) & ~(size_t)255;
    size_t off  = 0;
    float* dinv = base + off; off += npad;           // [N] deg -> dinv
    float* xw   = base + off; off += NF;             // [N*16] xw1 then xw2
    float* agg1 = base + off; off += NF;             // [N*16] agg1 -> x1
    float* agg2 = base + off; off += NF;             // [N*16] agg2 -> x2
    float* sums = base + off; off += (size_t)G * 16; // [G*16]
    float* cnts = base + off; off += G;              // [G]
    off = (off + 63) & ~(size_t)63;                  // 256B align for b128 loads
    __bf16* bp1 = (__bf16*)(base + off); off += 1024; // 2048 bf16 (K=128 pack)
    __bf16* bp2 = (__bf16*)(base + off); off += 256;  // 512 bf16  (K=16 pack)

    const int T = 256;
    const int gN  = (N + T - 1) / T;
    const int gE  = (E + T - 1) / T;
    const int gNF = (int)((NF + T - 1) / T);
    const int gE4 = (int)(((size_t)E * 4 + T - 1) / T);
    const int nTiles = (N + 15) / 16;
    const int gTiles = (nTiles + WAVES_PER_BLOCK - 1) / WAVES_PER_BLOCK;

    // weight packing (tiny, once per call) + degree/normalization
    k_pack_b<128><<<1, 128, 0, stream>>>(W1, bp1);
    k_pack_b<16><<<1, 32, 0, stream>>>(W2, bp2);
    k_init_deg<<<gN, T, 0, stream>>>(dinv, N);
    k_deg_edges<<<gE, T, 0, stream>>>(dst, dinv, E);
    k_rsqrt<<<gN, T, 0, stream>>>(dinv, N);

    // ---- layer 1 ----
    gcn_gemm_n16_wmma<128><<<gTiles, WAVES_PER_BLOCK * 32, 0, stream>>>(
        features, bp1, xw, N, nTiles);
    k_zero<<<gNF, T, 0, stream>>>(agg1, NF);
    k_edge_agg<<<gE4, T, 0, stream>>>((const float4*)xw, dinv, src, dst, agg1, E);
    k_finalize<<<gNF, T, 0, stream>>>(agg1, xw, dinv, b1, NF);   // x1 in agg1

    // ---- layer 2 ----
    gcn_gemm_n16_wmma<16><<<gTiles, WAVES_PER_BLOCK * 32, 0, stream>>>(
        agg1, bp2, xw, N, nTiles);
    k_zero<<<gNF, T, 0, stream>>>(agg2, NF);
    k_edge_agg<<<gE4, T, 0, stream>>>((const float4*)xw, dinv, src, dst, agg2, E);
    k_finalize<<<gNF, T, 0, stream>>>(agg2, xw, dinv, b2, NF);   // x2 in agg2

    // ---- pool + head ----
    k_zero<<<(int)(((size_t)G * 17 + T - 1) / T), T, 0, stream>>>(sums, (size_t)G * 17);
    k_pool<<<gNF, T, 0, stream>>>(agg2, batch, sums, cnts, NF);
    k_head<<<(G + T - 1) / T, T, 0, stream>>>(sums, cnts, Wf, bf_, Wo, bo,
                                              (float*)d_out, G);
}
